// GCN_68667937128872
// MI455X (gfx1250) — matrix operand order repaired and checked by reading, not compiled, not run on hardware
//
#include <hip/hip_runtime.h>

// ---------------------------------------------------------------------------
// Types for CDNA5 WMMA (wave32)
// ---------------------------------------------------------------------------
typedef __attribute__((ext_vector_type(16))) __bf16 v16bf;
typedef __attribute__((ext_vector_type(8)))  __bf16 v8bf;
typedef __attribute__((ext_vector_type(8)))  float  v8f;

#define DDIM 512              // feature dim everywhere (in/hidden/out)
#define LN_EPS 1e-5f

__device__ __forceinline__ __bf16 f2bf(float f) {
    unsigned u = __float_as_uint(f);
    unsigned r = (u + 0x7FFFu + ((u >> 16) & 1u)) >> 16;   // RNE
    unsigned short s = (unsigned short)r;
    return *reinterpret_cast<__bf16*>(&s);
}

// ---------------------------------------------------------------------------
// Elementwise helpers
// ---------------------------------------------------------------------------
__global__ void fill_f32(float* p, float v, int n) {
    int i = blockIdx.x * blockDim.x + threadIdx.x;
    if (i < n) p[i] = v;
}

__global__ void count_deg(const int* __restrict__ col, float* __restrict__ deg, int E) {
    int e = blockIdx.x * blockDim.x + threadIdx.x;
    if (e < E) atomicAdd(&deg[col[e]], 1.0f);
}

__global__ void rsqrt_inplace(float* p, int n) {
    int i = blockIdx.x * blockDim.x + threadIdx.x;
    if (i < n) p[i] = rsqrtf(p[i]);        // deg >= 1 always (self loops)
}

__global__ void cvt_f32_bf16(const float* __restrict__ in, __bf16* __restrict__ out, int n) {
    int i = blockIdx.x * blockDim.x + threadIdx.x;
    if (i < n) out[i] = f2bf(in[i]);
}

// acc += dinv[n]^2 * xw + (b + lb)   (self-loop contribution + both biases)
__global__ void fold_self_bias(float* __restrict__ acc, const float* __restrict__ xw,
                               const float* __restrict__ dinv,
                               const float* __restrict__ b, const float* __restrict__ lb,
                               int total) {
    int i = blockIdx.x * blockDim.x + threadIdx.x;
    if (i < total) {
        int n = i >> 9;           // /512
        int c = i & 511;
        float dv = dinv[n];
        acc[i] += dv * dv * xw[i] + b[c] + lb[c];
    }
}

// xf = xf + relu(in);  ob = bf16(xf)
__global__ void relu_add_bf16(const float* __restrict__ in, float* __restrict__ xf,
                              __bf16* __restrict__ ob, int total) {
    int i = blockIdx.x * blockDim.x + threadIdx.x;
    if (i < total) {
        float v = xf[i] + fmaxf(in[i], 0.0f);
        xf[i] = v;
        ob[i] = f2bf(v);
    }
}

// ---------------------------------------------------------------------------
// LayerNorm (one wave32 per 512-wide row), bf16 output for WMMA GEMM input
// ---------------------------------------------------------------------------
__global__ __launch_bounds__(256)
void ln_to_bf16(const float* __restrict__ x, const float* __restrict__ g,
                const float* __restrict__ be, __bf16* __restrict__ out, int Nn) {
    int wv   = (blockIdx.x * 256 + threadIdx.x) >> 5;
    int lane = threadIdx.x & 31;
    if (wv >= Nn) return;
    const float* xr = x + (size_t)wv * DDIM;
    float vals[16];
    float s = 0.f, s2 = 0.f;
#pragma unroll
    for (int i = 0; i < 4; ++i) {
        float4 v = *(const float4*)(xr + lane * 16 + i * 4);
        vals[i*4+0] = v.x; vals[i*4+1] = v.y; vals[i*4+2] = v.z; vals[i*4+3] = v.w;
        s  += v.x + v.y + v.z + v.w;
        s2 += v.x*v.x + v.y*v.y + v.z*v.z + v.w*v.w;
    }
#pragma unroll
    for (int off = 16; off > 0; off >>= 1) {
        s  += __shfl_xor(s,  off);
        s2 += __shfl_xor(s2, off);
    }
    float mean = s * (1.0f / DDIM);
    float var  = s2 * (1.0f / DDIM) - mean * mean;
    float rstd = rsqrtf(var + LN_EPS);
    __bf16* orow = out + (size_t)wv * DDIM;
#pragma unroll
    for (int i = 0; i < 16; ++i) {
        int c = lane * 16 + i;
        orow[c] = f2bf((vals[i] - mean) * rstd * g[c] + be[c]);
    }
}

// Fused: r = relu(in); store r (f32, JK accumulator init); out = bf16(LN(r))
__global__ __launch_bounds__(256)
void relu_ln_to_bf16(const float* __restrict__ in, float* __restrict__ store,
                     const float* __restrict__ g, const float* __restrict__ be,
                     __bf16* __restrict__ out, int Nn) {
    int wv   = (blockIdx.x * 256 + threadIdx.x) >> 5;
    int lane = threadIdx.x & 31;
    if (wv >= Nn) return;
    const float* xr = in + (size_t)wv * DDIM;
    float* sr = store + (size_t)wv * DDIM;
    float vals[16];
    float s = 0.f, s2 = 0.f;
#pragma unroll
    for (int i = 0; i < 4; ++i) {
        float4 v = *(const float4*)(xr + lane * 16 + i * 4);
        v.x = fmaxf(v.x, 0.f); v.y = fmaxf(v.y, 0.f);
        v.z = fmaxf(v.z, 0.f); v.w = fmaxf(v.w, 0.f);
        *(float4*)(sr + lane * 16 + i * 4) = v;
        vals[i*4+0] = v.x; vals[i*4+1] = v.y; vals[i*4+2] = v.z; vals[i*4+3] = v.w;
        s  += v.x + v.y + v.z + v.w;
        s2 += v.x*v.x + v.y*v.y + v.z*v.z + v.w*v.w;
    }
#pragma unroll
    for (int off = 16; off > 0; off >>= 1) {
        s  += __shfl_xor(s,  off);
        s2 += __shfl_xor(s2, off);
    }
    float mean = s * (1.0f / DDIM);
    float var  = s2 * (1.0f / DDIM) - mean * mean;
    float rstd = rsqrtf(var + LN_EPS);
    __bf16* orow = out + (size_t)wv * DDIM;
#pragma unroll
    for (int i = 0; i < 16; ++i) {
        int c = lane * 16 + i;
        orow[c] = f2bf((vals[i] - mean) * rstd * g[c] + be[c]);
    }
}

// ---------------------------------------------------------------------------
// Edge scatter: acc[col] += dinv[row]*dinv[col] * xw[row]   (one wave per edge,
// 16 floats per lane; gathers + atomics hit L2: the 102 MB matrix < 192 MB L2)
// ---------------------------------------------------------------------------
__global__ __launch_bounds__(256)
void scatter_edges(const int* __restrict__ rows, const int* __restrict__ cols,
                   const float* __restrict__ dinv, const float* __restrict__ xw,
                   float* __restrict__ acc, int E) {
    int e    = (blockIdx.x * 256 + threadIdx.x) >> 5;
    int lane = threadIdx.x & 31;
    if (e >= E) return;
    int r = rows[e], c = cols[e];
    float nrm = dinv[r] * dinv[c];
    const float* src = xw  + (size_t)r * DDIM + lane * 16;
    float*       dst = acc + (size_t)c * DDIM + lane * 16;
#pragma unroll
    for (int i = 0; i < 4; ++i) {
        float4 v = *(const float4*)(src + i * 4);
        atomicAdd(dst + i*4 + 0, nrm * v.x);
        atomicAdd(dst + i*4 + 1, nrm * v.y);
        atomicAdd(dst + i*4 + 2, nrm * v.z);
        atomicAdd(dst + i*4 + 3, nrm * v.w);
    }
}

// ---------------------------------------------------------------------------
// WMMA bf16 GEMM:  C[M,512] = A[M,512] @ op(B[512,512])  (+ bias)
//   TRANSB=0 : op(B)=B    (B row-major [K][N])
//   TRANSB=1 : op(B)=B^T  (B row-major [N][K], torch Linear weight)
// Block: 256 threads (8 wave32s). Tile 256x64, BK=32. Each wave: 32 rows x 64
// cols = 8 WMMAs/stage. Double-buffered LDS + register staging: next-stage
// global loads issue BEFORE the WMMAs, ds_stores land AFTER them, one barrier
// per stage. LDS row stride 40 halfwords (80B): 16B-aligned, conflict-free
// ds_read_b128.
// ---------------------------------------------------------------------------
#define BM 256
#define BN 64
#define BK 32
#define NSTAGE (DDIM / BK)
#define LSTR 40

__device__ __forceinline__ v16bf afrag_ld(const __bf16* rp, int half) {
    // A 16x32 bf16: lane half 0 -> K {0..7,16..23}; half 1 -> K {8..15,24..31}
    v8bf lo = *(const v8bf*)(rp + half * 8);
    v8bf hi = *(const v8bf*)(rp + 16 + half * 8);
    return __builtin_shufflevector(lo, hi, 0,1,2,3,4,5,6,7,8,9,10,11,12,13,14,15);
}
__device__ __forceinline__ v16bf bfrag_ld(const __bf16* rp, int half) {
    // B 32x16 bf16: lane half 0 -> K {0..7,16..23}; half 1 -> K {8..15,24..31}
    v8bf lo = *(const v8bf*)(rp + half * 8);
    v8bf hi = *(const v8bf*)(rp + 16 + half * 8);
    return __builtin_shufflevector(lo, hi, 0,1,2,3,4,5,6,7,8,9,10,11,12,13,14,15);
}

template <bool TRANSB, bool BIAS>
__global__ __launch_bounds__(256)
void gemm_bf16_wmma(const __bf16* __restrict__ A, const __bf16* __restrict__ B,
                    const float* __restrict__ bias, float* __restrict__ C, int M) {
    __shared__ __bf16 sA[2][BM * LSTR];   // 40960 B
    __shared__ __bf16 sB[2][BN * LSTR];   // 10240 B

    const int tid  = threadIdx.x;
    const int wave = tid >> 5;
    const int lane = tid & 31;
    const int half = lane >> 4;
    const int l16  = lane & 15;

    const int r0 = blockIdx.x * BM;
    const int n0 = blockIdx.y * BN;

    // A staging: 256x32 bf16 = 1024 x 16B chunks; thread covers 4 (q*64 rows apart)
    const int arow = tid >> 2;            // 0..63
    const int acg  = (tid & 3) << 3;      // 0,8,16,24
    // B staging
    const int btrow = tid >> 2;           // TRANSB: 0..63
    const int btcg  = (tid & 3) << 3;
    const int bk    = tid >> 3;           // !TRANSB: 0..31
    const int bng   = (tid & 7) << 3;     // 0..56

    v8f acc[2][4] = {};
    uint4 ra[4];
    uint4 rb;

    auto load_regs = [&](int s) {
        const int k0 = s * BK;
#pragma unroll
        for (int q = 0; q < 4; ++q) {
            int grow = r0 + arow + q * 64;
            uint4 v = {0u, 0u, 0u, 0u};
            if (grow < M) v = *(const uint4*)(A + (size_t)grow * DDIM + k0 + acg);
            ra[q] = v;
        }
        if (TRANSB) rb = *(const uint4*)(B + (size_t)(n0 + btrow) * DDIM + k0 + btcg);
        else        rb = *(const uint4*)(B + (size_t)(k0 + bk) * DDIM + n0 + bng);
    };
    auto store_lds = [&](int buf) {
#pragma unroll
        for (int q = 0; q < 4; ++q)
            *(uint4*)(&sA[buf][(arow + q * 64) * LSTR + acg]) = ra[q];
        if (TRANSB) {
            *(uint4*)(&sB[buf][btrow * LSTR + btcg]) = rb;
        } else {
            const __bf16* e = (const __bf16*)&rb;
#pragma unroll
            for (int i = 0; i < 8; ++i) sB[buf][(bng + i) * LSTR + bk] = e[i];
        }
    };

    load_regs(0);
    store_lds(0);
    __syncthreads();

    for (int s = 0; s < NSTAGE; ++s) {
        const int cur = s & 1;
        const bool more = (s + 1 < NSTAGE);
        if (more) load_regs(s + 1);            // global loads in flight over WMMAs
        if (s + 2 < NSTAGE)                    // CDNA5 speculative prefetch (L2)
            __builtin_prefetch(A + (size_t)(r0 + arow) * DDIM + (s + 2) * BK + acg, 0, 0);

        const __bf16* sa = sA[cur];
        const __bf16* sb = sB[cur];
        v16bf a0 = afrag_ld(sa + (wave * 32 + l16) * LSTR, half);
        v16bf a1 = afrag_ld(sa + (wave * 32 + 16 + l16) * LSTR, half);
#pragma unroll
        for (int j = 0; j < 4; ++j) {
            v16bf bf = bfrag_ld(sb + (j * 16 + l16) * LSTR, half);
            acc[0][j] = __builtin_amdgcn_wmma_f32_16x16x32_bf16(
                false, a0, false, bf, (short)0, acc[0][j], false, false);
            acc[1][j] = __builtin_amdgcn_wmma_f32_16x16x32_bf16(
                false, a1, false, bf, (short)0, acc[1][j], false, false);
        }
        if (more) store_lds(cur ^ 1);          // loadcnt wait lands after WMMAs
        __syncthreads();
    }

    // ---- store: VGPR v of lane -> row (strip*16 + 8*half + v), col l16 ----
    const int rowBase = r0 + wave * 32;
#pragma unroll
    for (int st = 0; st < 2; ++st) {
#pragma unroll
        for (int j = 0; j < 4; ++j) {
            int cc = n0 + j * 16 + l16;
            float badd = BIAS ? bias[cc] : 0.0f;
#pragma unroll
            for (int v = 0; v < 8; ++v) {
                int rr = rowBase + st * 16 + half * 8 + v;
                if (rr < M) C[(size_t)rr * DDIM + cc] = acc[st][j][v] + badd;
            }
        }
    }
}

// ---------------------------------------------------------------------------
// Orchestration
// ---------------------------------------------------------------------------
extern "C" void kernel_launch(void* const* d_in, const int* in_sizes, int n_in,
                              void* d_out, int out_size, void* d_ws, size_t ws_size,
                              hipStream_t stream) {
    const float* x    = (const float*)d_in[0];
    const int*   ei   = (const int*)d_in[1];
    const float* W0   = (const float*)d_in[2];
    const float* b0   = (const float*)d_in[3];
    const float* lw0  = (const float*)d_in[4];
    const float* lb0  = (const float*)d_in[5];
    const float* g0   = (const float*)d_in[6];
    const float* be0  = (const float*)d_in[7];
    const float* W1   = (const float*)d_in[8];
    const float* b1   = (const float*)d_in[9];
    const float* lw1  = (const float*)d_in[10];
    const float* lb1  = (const float*)d_in[11];
    const float* g1   = (const float*)d_in[12];
    const float* be1  = (const float*)d_in[13];
    const float* pw   = (const float*)d_in[14];
    const float* pb   = (const float*)d_in[15];

    const int N = in_sizes[0] / DDIM;       // 50000
    const int E = in_sizes[1] / 2;          // 400000
    const int* erow = ei;                   // sources
    const int* ecol = ei + E;               // targets
    const int total = N * DDIM;

    // -------- workspace carve (all 16B-aligned chunk sizes) --------
    char* p = (char*)d_ws;
    float* dinv = (float*)p;                 p += (size_t)N * 4;                 // deg -> dinv
    float* F0   = (float*)p;                 p += (size_t)total * 4;             // xw
    float* F1   = (float*)p;                 p += (size_t)total * 4;             // lin/acc
    __bf16* B0  = (__bf16*)p;                p += (size_t)total * 2;             // bf16 activations
    __bf16* W0b = (__bf16*)p;                p += (size_t)DDIM * DDIM * 2;
    __bf16* l0b = (__bf16*)p;                p += (size_t)DDIM * DDIM * 2;
    __bf16* W1b = (__bf16*)p;                p += (size_t)DDIM * DDIM * 2;
    __bf16* l1b = (__bf16*)p;                p += (size_t)DDIM * DDIM * 2;
    __bf16* pwb = (__bf16*)p;                p += (size_t)DDIM * DDIM * 2;
    float* F2 = (float*)d_out;               // JK accumulator lives in d_out

    const int TPB = 256;
    const int gN    = (N + TPB - 1) / TPB;
    const int gE    = (E + TPB - 1) / TPB;
    const int gTot  = (total + TPB - 1) / TPB;
    const int gRow  = (N + 7) / 8;                // wave-per-row kernels
    const int gEdge = (E + 7) / 8;                // wave-per-edge scatter
    const int gW    = (DDIM * DDIM + TPB - 1) / TPB;
    dim3 gGemm((N + BM - 1) / BM, DDIM / BN);

    // degree / dinv
    fill_f32<<<gN, TPB, 0, stream>>>(dinv, 1.0f, N);                 // self loops
    count_deg<<<gE, TPB, 0, stream>>>(ecol, dinv, E);
    rsqrt_inplace<<<gN, TPB, 0, stream>>>(dinv, N);

    // weights -> bf16
    cvt_f32_bf16<<<gW, TPB, 0, stream>>>(W0,  W0b, DDIM * DDIM);
    cvt_f32_bf16<<<gW, TPB, 0, stream>>>(lw0, l0b, DDIM * DDIM);
    cvt_f32_bf16<<<gW, TPB, 0, stream>>>(W1,  W1b, DDIM * DDIM);
    cvt_f32_bf16<<<gW, TPB, 0, stream>>>(lw1, l1b, DDIM * DDIM);
    cvt_f32_bf16<<<gW, TPB, 0, stream>>>(pw,  pwb, DDIM * DDIM);

    // ---- layer 0 ----
    ln_to_bf16<<<gRow, TPB, 0, stream>>>(x, g0, be0, B0, N);
    gemm_bf16_wmma<true,  false><<<gGemm, TPB, 0, stream>>>(B0, l0b, nullptr, F1, N); // lin
    gemm_bf16_wmma<false, false><<<gGemm, TPB, 0, stream>>>(B0, W0b, nullptr, F0, N); // xw
    fold_self_bias<<<gTot, TPB, 0, stream>>>(F1, F0, dinv, b0, lb0, total);
    scatter_edges<<<gEdge, TPB, 0, stream>>>(erow, ecol, dinv, F0, F1, E);
    relu_ln_to_bf16<<<gRow, TPB, 0, stream>>>(F1, F2, g1, be1, B0, N);   // F2 = relu(h0)

    // ---- layer 1 ----
    gemm_bf16_wmma<true,  false><<<gGemm, TPB, 0, stream>>>(B0, l1b, nullptr, F1, N);
    gemm_bf16_wmma<false, false><<<gGemm, TPB, 0, stream>>>(B0, W1b, nullptr, F0, N);
    fold_self_bias<<<gTot, TPB, 0, stream>>>(F1, F0, dinv, b1, lb1, total);
    scatter_edges<<<gEdge, TPB, 0, stream>>>(erow, ecol, dinv, F0, F1, E);
    relu_add_bf16<<<gTot, TPB, 0, stream>>>(F1, F2, B0, total);          // F2 += relu(h1)

    // ---- prediction head: d_out = xf @ pw^T + pb ----
    gemm_bf16_wmma<true, true><<<gGemm, TPB, 0, stream>>>(B0, pwb, pb, (float*)d_out, N);
}
